// EncoderLayer_26053271617912
// MI455X (gfx1250) — matrix operand order repaired and checked by reading, not compiled, not run on hardware
//
#include <hip/hip_runtime.h>

typedef __attribute__((ext_vector_type(16))) __bf16 v16bf;
typedef __attribute__((ext_vector_type(8)))  __bf16 v8bf;
typedef __attribute__((ext_vector_type(4)))  __bf16 v4bf;
typedef __attribute__((ext_vector_type(8)))  float  v8f;
typedef __attribute__((ext_vector_type(4)))  float  v4f;

#define D_MODEL 768
#define N_HEADS 12
#define D_HEAD  64
#define D_FF    3072
#define SEQ     2048
#define BATCH   2
#define TOKENS  (BATCH * SEQ)

__device__ __forceinline__ __bf16 f2bf(float f) {
  unsigned u = __builtin_bit_cast(unsigned, f);
  unsigned short r = (unsigned short)((u + 0x7FFFu + ((u >> 16) & 1u)) >> 16);
  return __builtin_bit_cast(__bf16, r);
}

__device__ __forceinline__ v16bf cat8(v8bf lo, v8bf hi) {
  v16bf r;
#pragma unroll
  for (int i = 0; i < 8; ++i) { r[i] = lo[i]; r[i + 8] = hi[i]; }
  return r;
}

__device__ __forceinline__ v8f wmma_bf16(v16bf a, v16bf b, v8f c) {
  return __builtin_amdgcn_wmma_f32_16x16x32_bf16(false, a, false, b,
                                                 (short)0, c, false, false);
}

// Async 16-byte global -> LDS copy (CDNA5 ASYNCcnt path, no VGPR round trip).
// Generic LDS pointers truncate to the 32-bit LDS offset (aperture rule).
__device__ __forceinline__ void async_ld16(const void* g, void* lds) {
  unsigned l = (unsigned)(size_t)lds;
  unsigned long long ga = (unsigned long long)(size_t)g;
  asm volatile("global_load_async_to_lds_b128 %0, %1, off"
               :: "v"(l), "v"(ga) : "memory");
}
__device__ __forceinline__ void wait_async0() {
  asm volatile("s_wait_asynccnt 0x0" ::: "memory");
}

// ---------------------------------------------------------------------------
// fp32 -> bf16 bulk convert (one-time for x and all weights).
// ---------------------------------------------------------------------------
__global__ void __launch_bounds__(256)
cvt_bf16_kernel(const float* __restrict__ in, __bf16* __restrict__ out, int n) {
  int i = (blockIdx.x * 256 + threadIdx.x) * 4;
  if (i < n) {
    v4f v = *(const v4f*)(in + i);
    v4bf o;
#pragma unroll
    for (int j = 0; j < 4; ++j) o[j] = f2bf(v[j]);
    *(v4bf*)(out + i) = o;
  }
}

// ---------------------------------------------------------------------------
// GEMM: C[M,N] = A[M,K] (bf16) @ W[N,K]^T (bf16) + bias[N] via bf16 WMMA.
// Block tile 128x128, BK=32, 256 threads = 8 waves, wave tile 32x64 (2x4 WMMA).
// Both operand tiles staged with async global->LDS copies.
// ---------------------------------------------------------------------------
template <bool RELU, bool OUT_BF16>
__global__ void __launch_bounds__(256)
gemm_kernel(const __bf16* __restrict__ A, const __bf16* __restrict__ W,
            const float* __restrict__ bias, void* __restrict__ Cout,
            int N, int K) {
  __shared__ __bf16 As[128 * 32];
  __shared__ __bf16 Bs[128 * 32];
  const int tid  = threadIdx.x;
  const int lane = tid & 31;
  const int wave = tid >> 5;
  const int m0 = blockIdx.y * 128;
  const int n0 = blockIdx.x * 128;
  const int wm = (wave & 3) * 32;   // 4 row groups of 32
  const int wn = (wave >> 2) * 64;  // 2 col groups of 64
  const int l15 = lane & 15;
  const bool lo16 = lane < 16;

  v8f acc[2][4];
#pragma unroll
  for (int mt = 0; mt < 2; ++mt)
#pragma unroll
    for (int nt = 0; nt < 4; ++nt)
#pragma unroll
      for (int j = 0; j < 8; ++j) acc[mt][nt][j] = 0.0f;

  for (int k0 = 0; k0 < K; k0 += 32) {
    // Stage A and W 128x32 bf16 tiles (512 groups of 8 elems each).
#pragma unroll
    for (int i = 0; i < 2; ++i) {
      int g = tid + i * 256;
      int row = g >> 2, col = (g & 3) << 3;
      async_ld16(&A[(size_t)(m0 + row) * K + k0 + col], &As[row * 32 + col]);
      async_ld16(&W[(size_t)(n0 + row) * K + k0 + col], &Bs[row * 32 + col]);
    }
    if (k0 + 32 < K) {  // hint next K-slab toward the caches
      __builtin_prefetch(A + (size_t)(m0 + (tid >> 1)) * K + k0 + 32 + (tid & 1) * 16, 0, 0);
      __builtin_prefetch(W + (size_t)(n0 + (tid >> 1)) * K + k0 + 32 + (tid & 1) * 16, 0, 0);
    }
    wait_async0();
    __syncthreads();

    // A fragment (16-bit 16x32 layout): lanes 0-15 hold K {0..7,16..23},
    // lanes 16-31 hold K {8..15,24..31} of row M=lane%16.
    v16bf afrag[2];
#pragma unroll
    for (int mt = 0; mt < 2; ++mt) {
      const __bf16* p = &As[(wm + mt * 16 + l15) * 32 + (lo16 ? 0 : 8)];
      afrag[mt] = cat8(*(const v8bf*)p, *(const v8bf*)(p + 16));
    }
    // B fragment (32x16): lanes 0-15 hold K {0..7,16..23}, lanes 16-31 K {8..15,24..31} of column n.
#pragma unroll
    for (int nt = 0; nt < 4; ++nt) {
      const __bf16* p = &Bs[(wn + nt * 16 + l15) * 32 + (lo16 ? 0 : 8)];
      v16bf bfrag = cat8(*(const v8bf*)p, *(const v8bf*)(p + 16));
#pragma unroll
      for (int mt = 0; mt < 2; ++mt)
        acc[mt][nt] = wmma_bf16(afrag[mt], bfrag, acc[mt][nt]);
    }
    __syncthreads();
  }

  // Epilogue: + bias, optional ReLU, fp32 or bf16 store.
#pragma unroll
  for (int mt = 0; mt < 2; ++mt)
#pragma unroll
    for (int nt = 0; nt < 4; ++nt) {
      int col = n0 + wn + nt * 16 + l15;
      float bcol = bias[col];
#pragma unroll
      for (int j = 0; j < 8; ++j) {
        int row = m0 + wm + mt * 16 + (lo16 ? j : j + 8);
        float v = acc[mt][nt][j] + bcol;
        if (RELU) v = fmaxf(v, 0.0f);
        if (OUT_BF16)
          ((__bf16*)Cout)[(size_t)row * N + col] = f2bf(v);
        else
          ((float*)Cout)[(size_t)row * N + col] = v;
      }
    }
}

// ---------------------------------------------------------------------------
// Flash attention: one block = one (b, h, 128 q-rows). 8 waves, each wave
// owns a 16-row strip. Scores and P@V both via bf16 WMMA, online softmax.
// Q/K/V bf16 [token, h*64+d]; output bf16 same layout (feeds out-proj GEMM).
// Q and K chunks staged with async global->LDS copies; V transposed via VGPRs.
// ---------------------------------------------------------------------------
__global__ void __launch_bounds__(256)
attn_kernel(const __bf16* __restrict__ Q, const __bf16* __restrict__ Km,
            const __bf16* __restrict__ Vm, const int* __restrict__ mask,
            __bf16* __restrict__ O) {
  __shared__ __bf16 Qs[128 * 64];   // [qrow][d]
  __shared__ __bf16 Ks[64 * 64];    // [krow][d]   (B operand of Q*K^T)
  __shared__ __bf16 Vt[64 * 64];    // [d][krow]   (B operand of P*V)
  __shared__ __bf16 Ps[8][16 * 64]; // per-wave P scratch [qrow][krow]

  const int tid = threadIdx.x, lane = tid & 31, wave = tid >> 5;
  const int l15 = lane & 15;
  const bool lo16 = lane < 16;
  const int bh = blockIdx.y, b = bh / N_HEADS, h = bh % N_HEADS;
  const int q0 = blockIdx.x * 128;
  const size_t base = (size_t)b * SEQ * D_MODEL + (size_t)h * D_HEAD;

  // Kick off the Q tile asynchronously (waited inside first loop iteration).
#pragma unroll
  for (int i = 0; i < 4; ++i) {
    int g = tid + i * 256;
    int row = g >> 3, col = (g & 7) << 3;
    async_ld16(&Q[base + (size_t)(q0 + row) * D_MODEL + col],
               &Qs[row * 64 + col]);
  }

  float m_i[8], l_i[8];
  v8f acc[4];
#pragma unroll
  for (int j = 0; j < 8; ++j) { m_i[j] = -1e30f; l_i[j] = 0.0f; }
#pragma unroll
  for (int dt = 0; dt < 4; ++dt)
#pragma unroll
    for (int j = 0; j < 8; ++j) acc[dt][j] = 0.0f;

  const int strip = wave * 16;

  for (int c = 0; c < SEQ / 64; ++c) {
    __syncthreads();  // previous iteration's compute done before overwrite
    // Stage K chunk row-major asynchronously; V chunk transposed via VGPRs.
#pragma unroll
    for (int i = 0; i < 2; ++i) {
      int g = tid + i * 256;
      int row = g >> 3, col = (g & 7) << 3;
      async_ld16(&Km[base + (size_t)(c * 64 + row) * D_MODEL + col],
                 &Ks[row * 64 + col]);
      v8bf v = *(const v8bf*)&Vm[base + (size_t)(c * 64 + row) * D_MODEL + col];
#pragma unroll
      for (int j = 0; j < 8; ++j) Vt[(col + j) * 64 + row] = v[j];
    }
    wait_async0();    // also covers the Q tile on the first iteration
    __syncthreads();

    // S = Q @ K^T for this wave's 16x64 strip (4 n-tiles x 2 k-steps).
    v8f s[4];
#pragma unroll
    for (int nt = 0; nt < 4; ++nt)
#pragma unroll
      for (int j = 0; j < 8; ++j) s[nt][j] = 0.0f;
#pragma unroll
    for (int ks = 0; ks < 2; ++ks) {
      const __bf16* pa = &Qs[(strip + l15) * 64 + ks * 32 + (lo16 ? 0 : 8)];
      v16bf a = cat8(*(const v8bf*)pa, *(const v8bf*)(pa + 16));
#pragma unroll
      for (int nt = 0; nt < 4; ++nt) {
        const __bf16* pb = &Ks[(nt * 16 + l15) * 64 + ks * 32 + (lo16 ? 0 : 8)];
        v16bf bf = cat8(*(const v8bf*)pb, *(const v8bf*)(pb + 16));
        s[nt] = wmma_bf16(a, bf, s[nt]);
      }
    }
    // Scale by 1/sqrt(64) and apply key-column mask (mask==true -> -1e9).
#pragma unroll
    for (int nt = 0; nt < 4; ++nt) {
      int kcol = c * 64 + nt * 16 + l15;
      bool msk = mask[b * SEQ + kcol] != 0;
#pragma unroll
      for (int j = 0; j < 8; ++j) {
        float v = s[nt][j] * 0.125f;
        s[nt][j] = msk ? -1e9f : v;
      }
    }
    // Online softmax: rows live across the 16-lane half (xor 1..8 reductions).
    float alpha[8];
#pragma unroll
    for (int j = 0; j < 8; ++j) {
      float t = fmaxf(fmaxf(s[0][j], s[1][j]), fmaxf(s[2][j], s[3][j]));
#pragma unroll
      for (int off = 8; off >= 1; off >>= 1) t = fmaxf(t, __shfl_xor(t, off, 32));
      float mn = fmaxf(m_i[j], t);
      alpha[j] = __expf(m_i[j] - mn);
      m_i[j] = mn;
    }
#pragma unroll
    for (int j = 0; j < 8; ++j) {
      float r = 0.0f;
#pragma unroll
      for (int nt = 0; nt < 4; ++nt) {
        float p = __expf(s[nt][j] - m_i[j]);
        s[nt][j] = p;
        r += p;
      }
#pragma unroll
      for (int off = 8; off >= 1; off >>= 1) r += __shfl_xor(r, off, 32);
      l_i[j] = l_i[j] * alpha[j] + r;
#pragma unroll
      for (int dt = 0; dt < 4; ++dt) acc[dt][j] *= alpha[j];
    }
    // Re-layout P into an A-fragmentable LDS tile (per-wave, in-order DS).
    __bf16* Pw = Ps[wave];
#pragma unroll
    for (int nt = 0; nt < 4; ++nt) {
      int colp = nt * 16 + l15;
#pragma unroll
      for (int j = 0; j < 8; ++j) {
        int rowp = lo16 ? j : j + 8;
        Pw[rowp * 64 + colp] = f2bf(s[nt][j]);
      }
    }
    asm volatile("s_wait_dscnt 0" ::: "memory");
    // O += P @ V  (4 d-tiles x 2 k-steps).
#pragma unroll
    for (int ks = 0; ks < 2; ++ks) {
      const __bf16* pa = &Pw[l15 * 64 + ks * 32 + (lo16 ? 0 : 8)];
      v16bf a = cat8(*(const v8bf*)pa, *(const v8bf*)(pa + 16));
#pragma unroll
      for (int dt = 0; dt < 4; ++dt) {
        const __bf16* pb = &Vt[(dt * 16 + l15) * 64 + ks * 32 + (lo16 ? 0 : 8)];
        v16bf bf = cat8(*(const v8bf*)pb, *(const v8bf*)(pb + 16));
        acc[dt] = wmma_bf16(a, bf, acc[dt]);
      }
    }
  }
  // Normalize and store bf16 (layout [token, h*64+d] => ready for out-proj).
#pragma unroll
  for (int j = 0; j < 8; ++j) {
    float l = l_i[j];
    float inv = l > 0.0f ? 1.0f / l : 0.0f;
    int row = q0 + strip + (lo16 ? j : j + 8);
#pragma unroll
    for (int dt = 0; dt < 4; ++dt) {
      int d = dt * 16 + l15;
      O[base + (size_t)row * D_MODEL + d] = f2bf(acc[dt][j] * inv);
    }
  }
}

// ---------------------------------------------------------------------------
// out = gamma * (x+y - mean) / (std_unbiased + eps) + beta, one row per block.
// Optionally also emits a bf16 copy for the next GEMM's A operand.
// ---------------------------------------------------------------------------
template <bool DUP_BF16>
__global__ void __launch_bounds__(256)
add_ln_kernel(const float* __restrict__ X, const float* __restrict__ Y,
              const float* __restrict__ gamma, const float* __restrict__ beta,
              float* __restrict__ out, __bf16* __restrict__ out_bf) {
  const int row = blockIdx.x;
  const int tid = threadIdx.x;
  __shared__ float ws1[8], ws2[8];
  float v[3];
  float s = 0.0f, s2 = 0.0f;
#pragma unroll
  for (int i = 0; i < 3; ++i) {
    size_t idx = (size_t)row * D_MODEL + tid + i * 256;
    float t = X[idx] + Y[idx];
    v[i] = t; s += t; s2 += t * t;
  }
#pragma unroll
  for (int off = 16; off >= 1; off >>= 1) {
    s  += __shfl_xor(s, off, 32);
    s2 += __shfl_xor(s2, off, 32);
  }
  if ((tid & 31) == 0) { ws1[tid >> 5] = s; ws2[tid >> 5] = s2; }
  __syncthreads();
  float S = 0.0f, S2 = 0.0f;
#pragma unroll
  for (int w = 0; w < 8; ++w) { S += ws1[w]; S2 += ws2[w]; }
  float mean = S * (1.0f / (float)D_MODEL);
  float var  = (S2 - (float)D_MODEL * mean * mean) * (1.0f / (float)(D_MODEL - 1));
  float inv  = 1.0f / (sqrtf(fmaxf(var, 0.0f)) + 1e-6f);
#pragma unroll
  for (int i = 0; i < 3; ++i) {
    int col = tid + i * 256;
    float r = gamma[col] * (v[i] - mean) * inv + beta[col];
    out[(size_t)row * D_MODEL + col] = r;
    if (DUP_BF16) out_bf[(size_t)row * D_MODEL + col] = f2bf(r);
  }
}

// ---------------------------------------------------------------------------
extern "C" void kernel_launch(void* const* d_in, const int* in_sizes, int n_in,
                              void* d_out, int out_size, void* d_ws, size_t ws_size,
                              hipStream_t stream) {
  const float* x   = (const float*)d_in[0];
  const int*   msk = (const int*)d_in[1];
  const float* wq  = (const float*)d_in[2];  const float* bq  = (const float*)d_in[3];
  const float* wk  = (const float*)d_in[4];  const float* bk  = (const float*)d_in[5];
  const float* wv  = (const float*)d_in[6];  const float* bv  = (const float*)d_in[7];
  const float* wo  = (const float*)d_in[8];  const float* bo  = (const float*)d_in[9];
  const float* w1  = (const float*)d_in[10]; const float* b1  = (const float*)d_in[11];
  const float* w2  = (const float*)d_in[12]; const float* b2  = (const float*)d_in[13];
  const float* g1  = (const float*)d_in[14]; const float* be1 = (const float*)d_in[15];
  const float* g2  = (const float*)d_in[16]; const float* be2 = (const float*)d_in[17];

  char* ws = (char*)d_ws;
  const size_t N_X  = (size_t)TOKENS * D_MODEL;       // 3,145,728
  const size_t N_W  = (size_t)D_MODEL * D_MODEL;      //   589,824
  const size_t N_WF = (size_t)D_FF * D_MODEL;         // 2,359,296
  const size_t N_FF = (size_t)TOKENS * D_FF;          // 12,582,912

  size_t off = 0;
  auto take = [&](size_t bytes) { char* r = ws + off; off += bytes; return r; };
  // Persistent bf16 operands (live across the whole launch).
  __bf16* xb  = (__bf16*)take(N_X  * 2);
  __bf16* wqb = (__bf16*)take(N_W  * 2);
  __bf16* wkb = (__bf16*)take(N_W  * 2);
  __bf16* wvb = (__bf16*)take(N_W  * 2);
  __bf16* wob = (__bf16*)take(N_W  * 2);
  __bf16* w1b = (__bf16*)take(N_WF * 2);
  __bf16* w2b = (__bf16*)take(N_WF * 2);
  // Pipeline intermediates with lifetime-based reuse (peak ~80 MB).
  __bf16* qb  = (__bf16*)take(N_X * 2);
  __bf16* kb  = (__bf16*)take(N_X * 2);
  __bf16* vb  = (__bf16*)take(N_X * 2);
  __bf16* abb = (__bf16*)take(N_X * 2);   // attention output (bf16)
  float*  pb  = (float*)qb;               // out-proj out; q/k dead by then
  float*  x1  = (float*)take(N_X * 4);    // LN1 output fp32
  __bf16* x1b = abb;                      // LN1 output bf16; abb dead
  __bf16* f1b = (__bf16*)take(N_FF * 2);  // FF hidden, ReLU'd, bf16
  float*  f2  = (float*)qb;               // FF output fp32; pb dead after LN1

  dim3 blk(256, 1, 1);
  dim3 gD(D_MODEL / 128, TOKENS / 128);   // (6, 32)
  dim3 gF(D_FF / 128,    TOKENS / 128);   // (24, 32)
  auto cvtg = [](size_t n) { return dim3((unsigned)((n / 4 + 255) / 256)); };

  // One-time fp32 -> bf16 conversions (x + all six weight matrices).
  cvt_bf16_kernel<<<cvtg(N_X),  blk, 0, stream>>>(x,  xb,  (int)N_X);
  cvt_bf16_kernel<<<cvtg(N_W),  blk, 0, stream>>>(wq, wqb, (int)N_W);
  cvt_bf16_kernel<<<cvtg(N_W),  blk, 0, stream>>>(wk, wkb, (int)N_W);
  cvt_bf16_kernel<<<cvtg(N_W),  blk, 0, stream>>>(wv, wvb, (int)N_W);
  cvt_bf16_kernel<<<cvtg(N_W),  blk, 0, stream>>>(wo, wob, (int)N_W);
  cvt_bf16_kernel<<<cvtg(N_WF), blk, 0, stream>>>(w1, w1b, (int)N_WF);
  cvt_bf16_kernel<<<cvtg(N_WF), blk, 0, stream>>>(w2, w2b, (int)N_WF);

  gemm_kernel<false, true ><<<gD, blk, 0, stream>>>(xb,  wqb, bq, qb, D_MODEL, D_MODEL);
  gemm_kernel<false, true ><<<gD, blk, 0, stream>>>(xb,  wkb, bk, kb, D_MODEL, D_MODEL);
  gemm_kernel<false, true ><<<gD, blk, 0, stream>>>(xb,  wvb, bv, vb, D_MODEL, D_MODEL);
  attn_kernel<<<dim3(SEQ / 128, BATCH * N_HEADS), blk, 0, stream>>>(qb, kb, vb, msk, abb);
  gemm_kernel<false, false><<<gD, blk, 0, stream>>>(abb, wob, bo, pb, D_MODEL, D_MODEL);
  add_ln_kernel<true ><<<TOKENS, blk, 0, stream>>>(x, pb, g1, be1, x1, x1b);
  gemm_kernel<true,  true ><<<gF, blk, 0, stream>>>(x1b, w1b, b1, f1b, D_FF, D_MODEL);
  gemm_kernel<false, false><<<gD, blk, 0, stream>>>(f1b, w2b, b2, f2, D_MODEL, D_FF);
  add_ln_kernel<false><<<TOKENS, blk, 0, stream>>>(x1, f2, g2, be2, (float*)d_out, nullptr);

  (void)in_sizes; (void)n_in; (void)out_size; (void)ws_size;
}
